// NonLocalBlock_60172491817333
// MI455X (gfx1250) — matrix operand order repaired and checked by reading, not compiled, not run on hardware
//
#include <hip/hip_runtime.h>
#include <hip/hip_bf16.h>

// ---------------- CDNA5 WMMA plumbing ----------------
typedef __attribute__((ext_vector_type(16))) __bf16 v16bf;
typedef __attribute__((ext_vector_type(8)))  float  v8f;

// A-operand fragment (16x32 bf16): lane M = lane%16,
// elems 0..15 -> K = klo+0..15 (klo = 0 for lanes 0-15, 16 for 16-31)
// p points at &row[klo]; contiguous 32B.
__device__ __forceinline__ v16bf ldA(const __bf16* p) {
    v16bf r;
    ((uint4*)&r)[0] = *(const uint4*)(p);
    ((uint4*)&r)[1] = *(const uint4*)(p + 8);
    return r;
}
// B-operand fragment (32x16 bf16): lane N = lane%16,
// elems 0..15 -> K = klo+0..15 (klo = 0 for lanes 0-15, 16 for 16-31) — contiguous 32B.
__device__ __forceinline__ v16bf ldB(const __bf16* p) {
    v16bf r;
    ((uint4*)&r)[0] = *(const uint4*)(p);
    ((uint4*)&r)[1] = *(const uint4*)(p + 8);
    return r;
}
__device__ __forceinline__ v8f WMMA_BF16(v16bf a, v16bf b, v8f c) {
    return __builtin_amdgcn_wmma_f32_16x16x32_bf16(false, a, false, b, (short)0, c, false, false);
}

#define BATCH 64
#define CCH   512
#define RCH   32
#define HW    1024

// ---------------- Kernel 0: weight prep ----------------
// Wmr = W_restore(512x32) @ W_mask(32x32); convert all weights to bf16.
__global__ __launch_bounds__(256) void k_prep(const float* __restrict__ Wreduce,
                                              const float* __restrict__ Wrestore,
                                              const float* __restrict__ Wphi,
                                              const float* __restrict__ Wtheta,
                                              const float* __restrict__ Wg,
                                              const float* __restrict__ Wmask,
                                              __bf16* __restrict__ wredb,
                                              __bf16* __restrict__ wphib,
                                              __bf16* __restrict__ wthetab,
                                              __bf16* __restrict__ wgb,
                                              __bf16* __restrict__ wmrb) {
    const int t = threadIdx.x;
    for (int i = t; i < CCH * RCH; i += 256) {
        int o = i >> 5, r = i & 31;
        float s = 0.f;
        #pragma unroll
        for (int k = 0; k < 32; ++k) s += Wrestore[o * 32 + k] * Wmask[k * 32 + r];
        wmrb[i] = (__bf16)s;
    }
    for (int i = t; i < RCH * CCH; i += 256) wredb[i] = (__bf16)Wreduce[i];
    for (int i = t; i < RCH * RCH; i += 256) {
        wphib[i]   = (__bf16)Wphi[i];
        wthetab[i] = (__bf16)Wtheta[i];
        wgb[i]     = (__bf16)Wg[i];
    }
}

// ---------------- Kernel 1: reduce + theta/phi/g projections ----------------
// grid (HW/64, B), 128 threads (4 waves), wave w owns 16 columns.
#define XSTR 40   // LDS row stride (bf16 elems): 80B rows, 16B aligned, conflict-friendly
__global__ __launch_bounds__(128) void k_proj(const float* __restrict__ x,
                                              const __bf16* __restrict__ wredb,
                                              const __bf16* __restrict__ wphib,
                                              const __bf16* __restrict__ wthetab,
                                              const __bf16* __restrict__ wgb,
                                              const float* __restrict__ posdec,
                                              __bf16* __restrict__ theta_ws,   // [B][HW][R]
                                              __bf16* __restrict__ phi_ws,     // [B][HW][R]
                                              __bf16* __restrict__ g_ws) {     // [B][R][HW]
    __shared__ __align__(16) __bf16 xT[64 * XSTR];   // x tile transposed: [n_local][k]
    __shared__ __align__(16) __bf16 xrT[64 * XSTR];  // xr tile transposed: [n_local][r]
    const int b    = blockIdx.y;
    const int n0   = blockIdx.x << 6;
    const int t    = threadIdx.x;
    const int lane = t & 31, w = t >> 5;
    const int lo   = lane & 15, hi = lane >> 4;
    const int abase = hi ? 8 : 0, klo = hi ? 16 : 0;
    const int nloc  = (w << 4) + lo;                 // this lane's column within the 64-wide tile
    const float* xb = x + (size_t)b * CCH * HW;

    v8f c0 = {}; v8f c1 = {};                        // xr rows 0-15 / 16-31 for our 16 cols
    for (int k0 = 0; k0 < CCH; k0 += 32) {
        // stage 32(k) x 64(n) f32 tile -> bf16 transposed in LDS
        {
            const int kk = t >> 2;
            const int cs = (t & 3) << 4;
            const float* src = xb + (size_t)(k0 + kk) * HW + n0 + cs;
            float tmp[16];
            ((float4*)tmp)[0] = ((const float4*)src)[0];
            ((float4*)tmp)[1] = ((const float4*)src)[1];
            ((float4*)tmp)[2] = ((const float4*)src)[2];
            ((float4*)tmp)[3] = ((const float4*)src)[3];
            #pragma unroll
            for (int i = 0; i < 16; ++i) xT[(cs + i) * XSTR + kk] = (__bf16)tmp[i];
        }
        __syncthreads();
        v16bf bx = ldB(&xT[nloc * XSTR + klo]);
        v16bf a0 = ldA(wredb + (size_t)(0  + lo) * CCH + k0 + klo);
        v16bf a1 = ldA(wredb + (size_t)(16 + lo) * CCH + k0 + klo);
        c0 = WMMA_BF16(a0, bx, c0);
        c1 = WMMA_BF16(a1, bx, c1);
        __syncthreads();
    }
    // xr (32 rows x 16 cols per wave) -> LDS transposed bf16 [n][r]
    #pragma unroll
    for (int v = 0; v < 8; ++v) {
        xrT[nloc * XSTR + (v + (hi << 3))]      = (__bf16)c0[v];
        xrT[nloc * XSTR + (16 + v + (hi << 3))] = (__bf16)c1[v];
    }
    __syncthreads();

    v16bf bxr = ldB(&xrT[nloc * XSTR + klo]);    // K = r_in (32)
    const size_t nro = ((size_t)b * HW + n0 + nloc) * RCH;  // [b][n][.] base for theta/phi

    // theta = W_theta @ xr, stored [n][r]
    {
        v16bf a0 = ldA(wthetab + (0  + lo) * RCH + klo);
        v16bf a1 = ldA(wthetab + (16 + lo) * RCH + klo);
        v8f z = {};
        v8f d0 = WMMA_BF16(a0, bxr, z);
        v8f d1 = WMMA_BF16(a1, bxr, z);
        #pragma unroll
        for (int v = 0; v < 8; ++v) {
            theta_ws[nro + (v + (hi << 3))]      = (__bf16)d0[v];
            theta_ws[nro + (16 + v + (hi << 3))] = (__bf16)d1[v];
        }
    }
    // phi = W_phi @ xr, stored [n][r]
    {
        v16bf a0 = ldA(wphib + (0  + lo) * RCH + klo);
        v16bf a1 = ldA(wphib + (16 + lo) * RCH + klo);
        v8f z = {};
        v8f d0 = WMMA_BF16(a0, bxr, z);
        v8f d1 = WMMA_BF16(a1, bxr, z);
        #pragma unroll
        for (int v = 0; v < 8; ++v) {
            phi_ws[nro + (v + (hi << 3))]      = (__bf16)d0[v];
            phi_ws[nro + (16 + v + (hi << 3))] = (__bf16)d1[v];
        }
    }
    // g = W_g @ xr + pos_dec, stored [r][m]
    {
        v16bf a0 = ldA(wgb + (0  + lo) * RCH + klo);
        v16bf a1 = ldA(wgb + (16 + lo) * RCH + klo);
        v8f z = {};
        v8f d0 = WMMA_BF16(a0, bxr, z);
        v8f d1 = WMMA_BF16(a1, bxr, z);
        const int ng = n0 + nloc;
        __bf16* gb = g_ws + (size_t)b * RCH * HW;
        #pragma unroll
        for (int v = 0; v < 8; ++v) {
            int r0v = v + (hi << 3);
            gb[(size_t)r0v * HW + ng]        = (__bf16)(d0[v] + posdec[(size_t)r0v * HW + ng]);
            gb[(size_t)(16 + r0v) * HW + ng] = (__bf16)(d1[v] + posdec[(size_t)(16 + r0v) * HW + ng]);
        }
    }
}

// ---------------- Kernel 2: fused attention (S -> softmax -> P@g) ----------------
// grid (HW/16, B), 128 threads. 16 attention rows per block; S row-block lives in LDS.
#define SPAD 1032   // f32 row stride: 4128B rows -> conflict-free lane fragments
__global__ __launch_bounds__(128) void k_attn(const __bf16* __restrict__ theta_ws,
                                              const __bf16* __restrict__ phi_ws,
                                              const __bf16* __restrict__ g_ws,
                                              const float* __restrict__ pos,   // [HW][HW]
                                              __bf16* __restrict__ y_ws) {     // [B][HW][R]
    extern __shared__ __align__(16) float Smem[];     // 16*SPAD floats (66KB, dynamic LDS)
    __bf16* Pb = (__bf16*)Smem;                       // P (bf16) aliases S; phase barriers keep it safe
    const int b    = blockIdx.y;
    const int n0   = blockIdx.x << 4;
    const int t    = threadIdx.x;
    const int lane = t & 31, w = t >> 5;
    const int lo   = lane & 15, hi = lane >> 4;
    const int abase = hi ? 8 : 0, klo = hi ? 16 : 0;
    const __bf16* thb = theta_ws + (size_t)b * HW * RCH;
    const __bf16* phb = phi_ws   + (size_t)b * HW * RCH;

    // ---- pass 1: S[n][m] = theta . phi  (K = R = 32, one WMMA per 16x16 tile)
    v16bf aT = ldA(thb + (size_t)(n0 + lo) * RCH + klo);
    for (int tm = 0; tm < 16; ++tm) {
        const int m0 = (w << 8) + (tm << 4);
        v16bf bP = ldB(phb + (size_t)(m0 + lo) * RCH + klo);
        v8f c = {};
        c = WMMA_BF16(aT, bP, c);
        #pragma unroll
        for (int v = 0; v < 8; ++v) Smem[(v + (hi << 3)) * SPAD + m0 + lo] = c[v];
    }
    __syncthreads();

    // ---- softmax rows (wave per row, 4 phases; barrier between reads of S and writes of P)
    for (int p = 0; p < 4; ++p) {
        const int rr = (p << 2) + w;
        float vals[32];
        float mx = -3.4e38f;
        #pragma unroll
        for (int j = 0; j < 32; ++j) {
            float v = Smem[rr * SPAD + lane + (j << 5)]
                    + pos[(size_t)(n0 + rr) * HW + lane + (j << 5)];
            vals[j] = v;
            mx = fmaxf(mx, v);
        }
        __syncthreads();   // all S reads of this phase done before Pb overwrites it
        #pragma unroll
        for (int off = 16; off; off >>= 1) mx = fmaxf(mx, __shfl_xor(mx, off, 32));
        float s = 0.f;
        #pragma unroll
        for (int j = 0; j < 32; ++j) { float e = __expf(vals[j] - mx); vals[j] = e; s += e; }
        #pragma unroll
        for (int off = 16; off; off >>= 1) s += __shfl_xor(s, off, 32);
        const float inv = 1.f / s;
        #pragma unroll
        for (int j = 0; j < 32; ++j)
            Pb[rr * SPAD + lane + (j << 5)] = (__bf16)(vals[j] * inv);
        __syncthreads();
    }

    // ---- pass 2: y[16x32] = P(16x1024) @ gT ; wave w: N-tile (w&1), K-half (w>>1)
    const int r0 = (w & 1) << 4;
    const int kh = (w >> 1) << 9;
    const __bf16* gb = g_ws + (size_t)b * RCH * HW;
    v8f c = {};
    for (int ks = 0; ks < 512; ks += 32) {
        const int mk = kh + ks;
        v16bf aP = ldA(&Pb[lo * SPAD + mk + klo]);
        v16bf bG = ldB(gb + (size_t)(r0 + lo) * HW + mk + klo);
        c = WMMA_BF16(aP, bG, c);
    }
    __syncthreads();
    float* part = Smem + 12 * SPAD;     // past Pb's 33KB footprint
    if (w >= 2) {
        float* pt = part + ((w - 2) << 8);
        #pragma unroll
        for (int v = 0; v < 8; ++v) pt[(v + (hi << 3)) * 16 + lo] = c[v];
    }
    __syncthreads();
    if (w < 2) {
        float* pt = part + (w << 8);
        __bf16* yb = y_ws + ((size_t)b * HW + n0) * RCH;
        #pragma unroll
        for (int v = 0; v < 8; ++v) {
            float r = c[v] + pt[(v + (hi << 3)) * 16 + lo];
            yb[(v + (hi << 3)) * RCH + r0 + lo] = (__bf16)r;
        }
    }
}

// ---------------- Kernel 3: out = (W_restore @ W_mask) @ y + x ----------------
// grid (HW/64, B), 128 threads; wave w owns 16 columns, loops 32 row-tiles (K=32, 1 WMMA each).
__global__ __launch_bounds__(128) void k_out(const float* __restrict__ x,
                                             const __bf16* __restrict__ wmrb,   // [512][32]
                                             const __bf16* __restrict__ y_ws,   // [B][HW][R]
                                             float* __restrict__ out) {
    const int b    = blockIdx.y;
    const int n0   = blockIdx.x << 6;
    const int t    = threadIdx.x;
    const int lane = t & 31, w = t >> 5;
    const int lo   = lane & 15, hi = lane >> 4;
    const int abase = hi ? 8 : 0, klo = hi ? 16 : 0;
    const int ncol  = n0 + (w << 4) + lo;
    v16bf bY = ldB(y_ws + ((size_t)b * HW + ncol) * RCH + klo);
    const float* xb = x   + (size_t)b * CCH * HW;
    float*       ob = out + (size_t)b * CCH * HW;
    for (int o0 = 0; o0 < CCH; o0 += 16) {
        v16bf aW = ldA(wmrb + (o0 + lo) * RCH + klo);
        v8f c = {};
        c = WMMA_BF16(aW, bY, c);
        #pragma unroll
        for (int v = 0; v < 8; ++v) {
            const size_t oo = (size_t)(o0 + v + (hi << 3)) * HW + ncol;
            ob[oo] = c[v] + xb[oo];
        }
    }
}

// ---------------- host launcher ----------------
extern "C" void kernel_launch(void* const* d_in, const int* in_sizes, int n_in,
                              void* d_out, int out_size, void* d_ws, size_t ws_size,
                              hipStream_t stream) {
    (void)in_sizes; (void)n_in; (void)out_size; (void)ws_size;
    const float* x        = (const float*)d_in[0];
    const float* Wreduce  = (const float*)d_in[1];
    const float* Wrestore = (const float*)d_in[2];
    const float* Wphi     = (const float*)d_in[3];
    const float* Wtheta   = (const float*)d_in[4];
    const float* Wg       = (const float*)d_in[5];
    const float* Wmask    = (const float*)d_in[6];
    const float* pos      = (const float*)d_in[7];
    const float* posdec   = (const float*)d_in[8];
    float* out = (float*)d_out;

    char* ws = (char*)d_ws;
    size_t off = 0;
    __bf16* wredb   = (__bf16*)(ws + off); off += (size_t)RCH * CCH * 2;        // 32 KB
    __bf16* wphib   = (__bf16*)(ws + off); off += (size_t)RCH * RCH * 2;        // 2 KB
    __bf16* wthetab = (__bf16*)(ws + off); off += (size_t)RCH * RCH * 2;
    __bf16* wgb     = (__bf16*)(ws + off); off += (size_t)RCH * RCH * 2;
    __bf16* wmrb    = (__bf16*)(ws + off); off += (size_t)CCH * RCH * 2;        // 32 KB
    off = (off + 255) & ~(size_t)255;
    __bf16* theta_b = (__bf16*)(ws + off); off += (size_t)BATCH * HW * RCH * 2; // 4 MB
    __bf16* phi_b   = (__bf16*)(ws + off); off += (size_t)BATCH * HW * RCH * 2; // 4 MB
    __bf16* g_b     = (__bf16*)(ws + off); off += (size_t)BATCH * RCH * HW * 2; // 4 MB
    __bf16* y_b     = (__bf16*)(ws + off); off += (size_t)BATCH * HW * RCH * 2; // 4 MB

    k_prep<<<1, 256, 0, stream>>>(Wreduce, Wrestore, Wphi, Wtheta, Wg, Wmask,
                                  wredb, wphib, wthetab, wgb, wmrb);
    k_proj<<<dim3(HW / 64, BATCH), 128, 0, stream>>>(x, wredb, wphib, wthetab, wgb,
                                                     posdec, theta_b, phi_b, g_b);
    k_attn<<<dim3(HW / 16, BATCH), 128, 16 * SPAD * sizeof(float), stream>>>(
        theta_b, phi_b, g_b, pos, y_b);
    k_out<<<dim3(HW / 64, BATCH), 128, 0, stream>>>(x, wmrb, y_b, out);
}
